// VGAE_68436008894705
// MI455X (gfx1250) — matrix operand order repaired and checked by reading, not compiled, not run on hardware
//
#include <hip/hip_runtime.h>
#include <hip/hip_bf16.h>

// ---------------------------------------------------------------------------
// VGAE forward for MI455X (gfx1250, wave32, WMMA + Tensor Data Mover).
// z (12000x192 bf16 = 4.6MB) is L2-resident; the z@z^T kernel stages 128x192
// panels into LDS with TENSOR_LOAD_TO_LDS (TDM pad feature de-conflicts LDS
// banks), then feeds v_wmma_f32_16x16x32_bf16 from ds_load_b128. Output is
// written with non-temporal stores (write-once 576MB stream).
// ---------------------------------------------------------------------------

typedef __attribute__((ext_vector_type(16))) __bf16 v16bf;
typedef __attribute__((ext_vector_type(8)))  __bf16 v8bf;
typedef __attribute__((ext_vector_type(8)))  float  v8f;
typedef __attribute__((ext_vector_type(4)))  unsigned int u32x4;
typedef __attribute__((ext_vector_type(8)))  int i32x8;
typedef __attribute__((ext_vector_type(4)))  int i32x4;

#define DEVI __device__ __forceinline__

static constexpr int kN = 12000;    // nodes
static constexpr long kE = 384000;  // edges
static constexpr int kF = 128;      // NFEAT
static constexpr int kD = 64;       // DIM
static constexpr int kH = 192;      // HID = DIM*L

#if __has_builtin(__builtin_amdgcn_tensor_load_to_lds)
#define USE_TDM 1
#else
#define USE_TDM 0
#endif

DEVI int lane_id() { return (int)(threadIdx.x & 31); }

DEVI v16bf cat8(v8bf lo, v8bf hi) {
  return __builtin_shufflevector(lo, hi, 0, 1, 2, 3, 4, 5, 6, 7,
                                 8, 9, 10, 11, 12, 13, 14, 15);
}

DEVI float fast_sigmoid(float x) {
  return __builtin_amdgcn_rcpf(1.f + __expf(-x));  // v_exp_f32 + v_rcp_f32
}

// A fragment (16x32 bf16) from fp32 row-major matrix, rows m0..m0+15, k0..k0+31.
// Per ISA 7.12.2: lanes 0-15 hold K={0..7,16..23}, lanes 16-31 K={8..15,24..31}.
DEVI v16bf a_frag_f32(const float* __restrict__ A, long lda, int m0, int k0) {
  int l = lane_id();
  int m = m0 + (l & 15);
  int kh = (l < 16) ? 0 : 8;
  const float* p = A + (long)m * lda + k0;
  v16bf r;
#pragma unroll
  for (int e = 0; e < 8; ++e) r[e] = (__bf16)p[kh + e];
#pragma unroll
  for (int e = 0; e < 8; ++e) r[8 + e] = (__bf16)p[16 + kh + e];
  return r;
}

// B fragment (32x16 bf16) from fp32 weight W[K x N] row-major, cols n0..n0+15.
// Lanes 0-15 hold K={0..7,16..23}, lanes 16-31 hold K={8..15,24..31}.
DEVI v16bf b_frag_w(const float* __restrict__ W, long ldw, int n0, int k0) {
  int l = lane_id();
  int n = n0 + (l & 15);
  int kb = (l < 16) ? 0 : 8;
  v16bf r;
#pragma unroll
  for (int e = 0; e < 16; ++e) r[e] = (__bf16)W[(long)(k0 + kb + (e & 7) + (e >> 3) * 16) * ldw + n];
  return r;
}

// ------------------------- generic fused GEMM ------------------------------
// C[M,Nc] = epilogue(A[M,K] @ W[K,Nc] + bias)  (+ optional residual add)
// block = 256 threads = 8 waves in a 4x2 grid of 16x16 tiles (64x32 per block)
template <int ACT, bool HAS_ADD>
__global__ __launch_bounds__(256) void gemm_bf16_kernel(
    const float* __restrict__ A, int lda, const float* __restrict__ W, int ldw,
    const float* __restrict__ bias, const float* __restrict__ add, int ldadd,
    float* __restrict__ C, int ldc, int M, int Nc, int K) {
  int w = (int)(threadIdx.x >> 5);
  int bm = (int)blockIdx.x * 64 + (w & 3) * 16;
  int bn = (int)blockIdx.y * 32 + (w >> 2) * 16;
  if (bm >= M) return;  // M is a multiple of 16 -> whole-tile guard is exact
  v8f acc = {};
  for (int k0 = 0; k0 < K; k0 += 32) {
    v16bf a = a_frag_f32(A, lda, bm, k0);
    v16bf b = b_frag_w(W, ldw, bn, k0);
    acc = __builtin_amdgcn_wmma_f32_16x16x32_bf16(false, a, false, b,
                                                  (short)0, acc, false, false);
  }
  int l = lane_id();
  int n = bn + (l & 15);
  int mr0 = bm + ((l < 16) ? 0 : 8);
  float bv = bias[n];
#pragma unroll
  for (int r = 0; r < 8; ++r) {
    int m = mr0 + r;
    float v = acc[r] + bv;
    if (ACT == 1) v = v > 0.f ? v : 0.01f * v;  // leaky_relu(0.01)
    if (HAS_ADD) v += add[(long)m * ldadd + n];
    C[(long)m * ldc + n] = v;
  }
}

// ---------------------- graph aggregation (GIN, eps=0) ---------------------
__global__ void copy_ld_kernel(const float* __restrict__ src, int lds,
                               float* __restrict__ dst, int ldd, int M, int wdt) {
  long i = (long)blockIdx.x * blockDim.x + threadIdx.x;
  if (i >= (long)M * wdt) return;
  long n = i / wdt; int f = (int)(i % wdt);
  dst[n * ldd + f] = src[n * lds + f];
}

__global__ void scatter_add_kernel(const float* __restrict__ x, int ldx,
                                   const long long* __restrict__ esrc,
                                   const long long* __restrict__ edst,
                                   float* __restrict__ h, int ldh,
                                   long E, int wdt) {
  long i = (long)blockIdx.x * blockDim.x + threadIdx.x;
  if (i >= E * wdt) return;
  long e = i / wdt; int f = (int)(i % wdt);
  long s = (long)esrc[e], d = (long)edst[e];
  atomicAdd(&h[d * ldh + f], x[s * ldx + f]);  // global_atomic_add_f32
}

// ---------------------------- batch-norm stats -----------------------------
__global__ void colstats_kernel(const float* __restrict__ h, int ldh, int M,
                                float* __restrict__ mu, float* __restrict__ var) {
  int c = (int)blockIdx.x;
  float s = 0.f, sq = 0.f;
  for (int r = (int)threadIdx.x; r < M; r += (int)blockDim.x) {
    float v = h[(long)r * ldh + c];
    s += v; sq += v * v;
  }
  __shared__ float ss[256], sqq[256];
  int t = (int)threadIdx.x;
  ss[t] = s; sqq[t] = sq;
  __syncthreads();
  for (int o = 128; o > 0; o >>= 1) {
    if (t < o) { ss[t] += ss[t + o]; sqq[t] += sqq[t + o]; }
    __syncthreads();
  }
  if (t == 0) {
    float m = ss[0] / (float)M;
    mu[c] = m;
    var[c] = sqq[0] / (float)M - m * m;  // biased variance (torch BN norm)
  }
}

__global__ void bn_apply_kernel(const float* __restrict__ h, int ldh,
                                const float* __restrict__ mu,
                                const float* __restrict__ var,
                                const float* __restrict__ gamma,
                                const float* __restrict__ beta,
                                float* __restrict__ xcur, int ldx,
                                float* __restrict__ xcat, int ldcat, int coff,
                                int M, int C) {
  long i = (long)blockIdx.x * blockDim.x + threadIdx.x;
  if (i >= (long)M * C) return;
  long n = i / C; int c = (int)(i % C);
  float v = (h[n * ldh + c] - mu[c]) * rsqrtf(var[c] + 1e-4f) * gamma[c] + beta[c];
  xcur[n * ldx + c] = v;
  xcat[n * ldcat + coff + c] = v;
}

// ----------------------- z = noise*exp(logstd)+mean ------------------------
__global__ void z_kernel(const float* __restrict__ noise,
                         const float* __restrict__ mean,
                         const float* __restrict__ logstd,
                         __bf16* __restrict__ zb, long n) {
  long i = (long)blockIdx.x * blockDim.x + threadIdx.x;
  if (i >= n) return;
  zb[i] = (__bf16)(noise[i] * __expf(logstd[i]) + mean[i]);
}

// ------------------------- A_pred = sigmoid(z z^T) -------------------------
// Block tile 128x128; two 128x192 bf16 panels of z staged into LDS by the TDM
// (one tensor_load_to_lds per panel). TDM padding: after every 32 DWORDs
// stored, skip 4 DWORDs -> padded row stride 432B; 44*r mod 64 visits all 16
// bank groups so 16B fragment reads are conflict-free and 16B-aligned.
static constexpr int kTileRows  = 128;
static constexpr int kRowStride = 432;                      // 96 DW + 3*4 DW pads
static constexpr int kTileBytes = kTileRows * kRowStride;   // 55296

DEVI int lds_off(int row, int k) {   // byte offset of element (row, k) in padded tile
  return row * kRowStride + (k >> 6) * 144 + (k & 63) * 2;  // 128B chunk + 16B pad
}

#if USE_TDM
DEVI void tdm_load_tile(const __bf16* gbase, int row0, int rows, unsigned lds_byte_off) {
  unsigned long long ga =
      (unsigned long long)(__SIZE_TYPE__)gbase + (unsigned long long)row0 * (kH * 2);
  u32x4 g0;
  g0[0] = 1u;                                  // count=1, user descriptor
  g0[1] = lds_byte_off;                        // lds_addr (bytes)
  g0[2] = (unsigned)(ga & 0xFFFFFFFFu);        // global_addr[31:0]
  g0[3] = (unsigned)((ga >> 32) & 0x1FFFFFFu)  // global_addr[56:32]
          | (2u << 30);                        // type = 2 ("image")
  i32x8 g1;
  g1[0] = (1 << 16)      // data_size = 1 -> 2-byte elements
        | (1 << 20)      // pad_enable
        | (4 << 22)      // pad_interval code 4 -> every 32 DWORDs
        | (3 << 25);     // pad_amount code 3 -> 4 DWORDs
  g1[1] = (kH << 16);    // tensor_dim0 = 192 (bits 79:48, low half)
  g1[2] = (kN << 16);    // tensor_dim1 = 12000 (bits 111:80, low half)
  g1[3] = (kH << 16);    // tile_dim0 = 192 (bits 127:112)
  g1[4] = rows;          // tile_dim1 (bits 143:128); tile_dim2 = 0
  g1[5] = kH;            // tensor_dim0_stride = 192 (bits 207:160, low 32)
  g1[6] = 0;
  g1[7] = 0;
  i32x4 zr4 = {0, 0, 0, 0};                    // groups 2/3 unused (2D tensor)
  i32x8 zr8 = {0, 0, 0, 0, 0, 0, 0, 0};        // extra group (clang-23 6-arg form)
  __builtin_amdgcn_tensor_load_to_lds(g0, g1, zr4, zr4, zr8, 0);
}
#endif

DEVI v16bf a_frag_lds(const char* tile, int r0, int k0) {
  int l = lane_id();
  int row = r0 + (l & 15);
  int k = k0 + ((l < 16) ? 0 : 8);
  v8bf lo = *(const v8bf*)(tile + lds_off(row, k));
  v8bf hi = *(const v8bf*)(tile + lds_off(row, k + 16));
  return cat8(lo, hi);
}

DEVI v16bf b_frag_lds(const char* tile, int r0, int k0) {
  int l = lane_id();
  int row = r0 + (l & 15);
  int k = k0 + ((l < 16) ? 0 : 8);
  v8bf lo = *(const v8bf*)(tile + lds_off(row, k));
  v8bf hi = *(const v8bf*)(tile + lds_off(row, k + 16));
  return cat8(lo, hi);
}

__global__ __launch_bounds__(256) void adj_sigmoid_kernel(
    const __bf16* __restrict__ zb, float* __restrict__ out, int Nn, int K) {
  extern __shared__ char smem[];               // dynamic LDS, base offset 0
  char* tileA = smem;
  char* tileB = smem + kTileBytes;
  int row0A = (int)blockIdx.x * 128;
  int row0B = (int)blockIdx.y * 128;
  int rowsA = (Nn - row0A) < 128 ? (Nn - row0A) : 128;  // multiple of 16
  int rowsB = (Nn - row0B) < 128 ? (Nn - row0B) : 128;

#if USE_TDM
  if ((threadIdx.x >> 5) == 0) {               // one wave issues both DMAs
    tdm_load_tile(zb, row0A, rowsA, 0u);
    tdm_load_tile(zb, row0B, rowsB, (unsigned)kTileBytes);
    __builtin_amdgcn_s_wait_tensorcnt(0);
  }
  __syncthreads();
#else
  for (int i = (int)threadIdx.x; i < kTileRows * kH; i += (int)blockDim.x) {
    int r = i / kH, k = i % kH;
    __bf16 va = (r < rowsA) ? zb[(long)(row0A + r) * K + k] : (__bf16)0.f;
    __bf16 vb = (r < rowsB) ? zb[(long)(row0B + r) * K + k] : (__bf16)0.f;
    *(__bf16*)(tileA + lds_off(r, k)) = va;
    *(__bf16*)(tileB + lds_off(r, k)) = vb;
  }
  __syncthreads();
#endif

  int w = (int)(threadIdx.x >> 5);
  int wm = (w & 1) * 64;       // local row base (2 groups of 64 rows)
  int wn = (w >> 1) * 32;      // local col base (4 groups of 32 cols)
  int l = lane_id();
  v8f acc[4][2] = {};
  for (int k0 = 0; k0 < K; k0 += 32) {
    v16bf a[4], b[2];
#pragma unroll
    for (int i = 0; i < 4; ++i) a[i] = a_frag_lds(tileA, wm + i * 16, k0);
#pragma unroll
    for (int j = 0; j < 2; ++j) b[j] = b_frag_lds(tileB, wn + j * 16, k0);
#pragma unroll
    for (int i = 0; i < 4; ++i)
#pragma unroll
      for (int j = 0; j < 2; ++j)
        acc[i][j] = __builtin_amdgcn_wmma_f32_16x16x32_bf16(
            false, a[i], false, b[j], (short)0, acc[i][j], false, false);
  }
  // Epilogue: every 16x16 tile is uniformly in-range or out-of-range
  // (valid row/col counts are multiples of 16), so guards are scalar branches
  // and lanes never diverge around the stores.
#pragma unroll
  for (int i = 0; i < 4; ++i) {
    int tm = row0A + wm + i * 16;
    if (tm >= Nn) continue;                    // wave-uniform
    int mr0 = tm + ((l < 16) ? 0 : 8);
#pragma unroll
    for (int j = 0; j < 2; ++j) {
      int tn = row0B + wn + j * 16;
      if (tn >= Nn) continue;                  // wave-uniform
      int n = tn + (l & 15);
#pragma unroll
      for (int r = 0; r < 8; ++r) {
        float v = fast_sigmoid(acc[i][j][r]);
        __builtin_nontemporal_store(v, out + (long)(mr0 + r) * Nn + n);
      }
    }
  }
}

// ---------------------------------------------------------------------------
// host-side orchestration
// ---------------------------------------------------------------------------
struct EncLayer { const float *W1, *b1, *W2, *b2, *gamma, *beta; };
struct FFParams { const float *W[4], *b[4]; };

static void launch_gemm(const float* A, int lda, const float* W, int ldw,
                        const float* bias, const float* add, int ldadd,
                        float* C, int ldc, int M, int Nc, int K, int act,
                        hipStream_t s) {
  dim3 grid((M + 63) / 64, Nc / 32);
  if (add)
    gemm_bf16_kernel<0, true><<<grid, 256, 0, s>>>(A, lda, W, ldw, bias, add,
                                                   ldadd, C, ldc, M, Nc, K);
  else if (act == 1)
    gemm_bf16_kernel<1, false><<<grid, 256, 0, s>>>(A, lda, W, ldw, bias, nullptr,
                                                    0, C, ldc, M, Nc, K);
  else
    gemm_bf16_kernel<0, false><<<grid, 256, 0, s>>>(A, lda, W, ldw, bias, nullptr,
                                                    0, C, ldc, M, Nc, K);
}

static void run_encoder(const EncLayer* lyr, const float* input, int in_ld, int fin,
                        const long long* esrc, const long long* edst,
                        float* bufH, float* bufT, float* bufP, float* xcur,
                        float* xcat, float* stats, hipStream_t s) {
  const int M = kN;
  for (int li = 0; li < 3; ++li) {
    const float* cur = (li == 0) ? input : xcur;
    int ld = (li == 0) ? in_ld : kH;
    int wdt = (li == 0) ? fin : kD;
    long tot = (long)M * wdt;
    copy_ld_kernel<<<(unsigned)((tot + 255) / 256), 256, 0, s>>>(cur, ld, bufH, kH, M, wdt);
    long etot = kE * wdt;
    scatter_add_kernel<<<(unsigned)((etot + 255) / 256), 256, 0, s>>>(
        cur, ld, esrc, edst, bufH, kH, kE, wdt);
    // t = leaky_relu(h @ W1 + b1)
    launch_gemm(bufH, kH, lyr[li].W1, kD, lyr[li].b1, nullptr, 0,
                bufT, kD, M, kD, wdt, 1, s);
    // p = t @ W2 + b2
    launch_gemm(bufT, kD, lyr[li].W2, kD, lyr[li].b2, nullptr, 0,
                bufP, kD, M, kD, kD, 0, s);
    colstats_kernel<<<kD, 256, 0, s>>>(bufP, kD, M, stats, stats + kD);
    long btot = (long)M * kD;
    bn_apply_kernel<<<(unsigned)((btot + 255) / 256), 256, 0, s>>>(
        bufP, kD, stats, stats + kD, lyr[li].gamma, lyr[li].beta,
        xcur, kH, xcat, kH, kD * li, M, kD);
  }
}

static void run_ff(const FFParams& p, const float* xin, float* out,
                   float* tmp1, float* tmp2, hipStream_t s) {
  launch_gemm(xin, kH, p.W[0], kH, p.b[0], nullptr, 0, tmp1, kH, kN, kH, kH, 1, s);
  launch_gemm(tmp1, kH, p.W[1], kH, p.b[1], nullptr, 0, tmp2, kH, kN, kH, kH, 1, s);
  launch_gemm(tmp2, kH, p.W[2], kH, p.b[2], nullptr, 0, tmp1, kH, kN, kH, kH, 1, s);
  // out = h3 + xin @ Ws + bs
  launch_gemm(xin, kH, p.W[3], kH, p.b[3], tmp1, kH, out, kH, kN, kH, kH, 0, s);
}

extern "C" void kernel_launch(void* const* d_in, const int* in_sizes, int n_in,
                              void* d_out, int out_size, void* d_ws, size_t ws_size,
                              hipStream_t stream) {
  (void)in_sizes; (void)n_in; (void)out_size; (void)ws_size;
  int idx = 0;
  auto F = [&](void) { return (const float*)d_in[idx++]; };
  const float* x = F();
  const float* noise = F();
  EncLayer encB[3], encM[3], encL[3];
  EncLayer* encs[3] = {encB, encM, encL};
  for (int e = 0; e < 3; ++e)
    for (int li = 0; li < 3; ++li) {
      encs[e][li].W1 = F(); encs[e][li].b1 = F();
      encs[e][li].W2 = F(); encs[e][li].b2 = F();
      encs[e][li].gamma = F(); encs[e][li].beta = F();
    }
  FFParams ffB, ffM, ffL;
  FFParams* ffs[3] = {&ffB, &ffM, &ffL};
  for (int f = 0; f < 3; ++f)
    for (int i = 0; i < 4; ++i) { ffs[f]->W[i] = F(); ffs[f]->b[i] = F(); }
  const long long* eidx = (const long long*)d_in[idx++];
  const long long* esrc = eidx;
  const long long* edst = eidx + kE;
  idx++;  // batch: unused by the reference

  // workspace carve-up
  char* ws = (char*)d_ws;
  auto take = [&](size_t bytes) { char* p = ws; ws += (bytes + 255) & ~255ull; return p; };
  float* bufH  = (float*)take((size_t)kN * kH * 4);
  float* bufT  = (float*)take((size_t)kN * kD * 4);
  float* bufP  = (float*)take((size_t)kN * kD * 4);
  float* xcur  = (float*)take((size_t)kN * kH * 4);
  float* xcatA = (float*)take((size_t)kN * kH * 4);
  float* hidden= (float*)take((size_t)kN * kH * 4);
  float* xcatB = (float*)take((size_t)kN * kH * 4);
  float* tmp1  = (float*)take((size_t)kN * kH * 4);
  float* tmp2  = (float*)take((size_t)kN * kH * 4);
  float* meanO = (float*)take((size_t)kN * kH * 4);
  float* logsO = (float*)take((size_t)kN * kH * 4);
  float* stats = (float*)take((size_t)2 * kD * 4);
  __bf16* zb   = (__bf16*)take((size_t)kN * kH * 2);

  // hidden = ff_base(encoder(base, x))
  run_encoder(encB, x, kF, kF, esrc, edst, bufH, bufT, bufP, xcur, xcatA, stats, stream);
  run_ff(ffB, xcatA, hidden, tmp1, tmp2, stream);
  // mean = ff_mean(encoder(mean, hidden))
  run_encoder(encM, hidden, kH, kH, esrc, edst, bufH, bufT, bufP, xcur, xcatB, stats, stream);
  run_ff(ffM, xcatB, meanO, tmp1, tmp2, stream);
  // logstd = ff_logstd(encoder(logstd, hidden))
  run_encoder(encL, hidden, kH, kH, esrc, edst, bufH, bufT, bufP, xcur, xcatB, stats, stream);
  run_ff(ffL, xcatB, logsO, tmp1, tmp2, stream);

  // z (bf16) then A_pred = sigmoid(z z^T)
  long zn = (long)kN * kH;
  z_kernel<<<(unsigned)((zn + 255) / 256), 256, 0, stream>>>(noise, meanO, logsO, zb, zn);
  dim3 agrid((kN + 127) / 128, (kN + 127) / 128);
  adj_sigmoid_kernel<<<agrid, 256, 2 * kTileBytes, stream>>>(zb, (float*)d_out, kN, kH);
}
